// mobilnet_roifeatures_crossattention_83141976916041
// MI455X (gfx1250) — matrix operand order repaired and checked by reading, not compiled, not run on hardware
//
#include <hip/hip_runtime.h>
#include <math.h>

// Problem dims (fixed by the reference)
#define B_   64
#define R_   49
#define H_   1024
#define V_   10000
#define T_   32
#define H4_  4096
#define H2_  2048

typedef unsigned short bf16s;                      // bf16 storage
typedef __bf16 bf16t;
typedef __attribute__((ext_vector_type(16))) bf16t v16bf;
typedef __attribute__((ext_vector_type(8)))  float v8f;

union FragU { v16bf v; uint4 q[2]; };

__device__ __forceinline__ bf16s f2bf(float f) {
  unsigned u = __float_as_uint(f);
  u += 0x7FFFu + ((u >> 16) & 1u);                 // round-to-nearest-even
  return (bf16s)(u >> 16);
}
__device__ __forceinline__ float bf2f(bf16s h) {
  return __uint_as_float(((unsigned)h) << 16);
}
__device__ __forceinline__ float sigmoidf_(float x) { return 1.f / (1.f + expf(-x)); }

// A fragment: 16x32 (MxK) bf16. lane L: m = m0 + (L&15), group g = L>>4.
// VGPR0..3 hold K = kk+8g .. kk+8g+7 ; VGPR4..7 hold K = kk+16+8g .. +7.
__device__ __forceinline__ v16bf load_frag_a(const bf16s* A, int lda, int m, int kk, int g) {
  const bf16s* p = A + (size_t)m * lda + kk;
  FragU f;
  f.q[0] = *(const uint4*)(p + 8 * g);
  f.q[1] = *(const uint4*)(p + 16 + 8 * g);
  return f.v;
}
// B fragment: 32x16 (KxN), B[k][n] = W[n][k] (W row-major, row length = K).
// lane L: n = n0 + (L&15), g = L>>4, halves 0..7: K = kk+8g+i ; 8..15: K = kk+16+8g+(i-8).
__device__ __forceinline__ v16bf load_frag_b(const bf16s* W, int ldw, int n, int kk, int g) {
  const bf16s* p = W + (size_t)n * ldw + kk;
  FragU f;
  f.q[0] = *(const uint4*)(p + 8 * g);
  f.q[1] = *(const uint4*)(p + 16 + 8 * g);
  return f.v;
}

// One pipeline stage: A fragment + 4 B fragments (one 16x64 slab of K=32)
struct FragSet { v16bf a, b0, b1, b2, b3; };

__device__ __forceinline__ FragSet load_set(const bf16s* A, const bf16s* B, int K,
                                            int ma, const int* nc, int kk, int g) {
  FragSet f;
  f.a  = load_frag_a(A, K, ma, kk, g);
  f.b0 = load_frag_b(B, K, nc[0], kk, g);
  f.b1 = load_frag_b(B, K, nc[1], kk, g);
  f.b2 = load_frag_b(B, K, nc[2], kk, g);
  f.b3 = load_frag_b(B, K, nc[3], kk, g);
  return f;
}

// WGP-scope prefetch of the fragment rows for a future K step (pulls into
// L0/WGP$; no LOADcnt cost). Each global_prefetch_b8 fetches the cacheline.
__device__ __forceinline__ void prefetch_set(const bf16s* A, const bf16s* B, int K,
                                             int ma, const int* nc, int kk) {
  __builtin_prefetch(A + (size_t)ma    * K + kk, 0, 1);
  __builtin_prefetch(B + (size_t)nc[0] * K + kk, 0, 1);
  __builtin_prefetch(B + (size_t)nc[1] * K + kk, 0, 1);
  __builtin_prefetch(B + (size_t)nc[2] * K + kk, 0, 1);
  __builtin_prefetch(B + (size_t)nc[3] * K + kk, 0, 1);
}

__device__ __forceinline__ void mma_set(v8f* acc, const FragSet& f) {
  acc[0] = __builtin_amdgcn_wmma_f32_16x16x32_bf16(false, f.a, false, f.b0, (short)0, acc[0], false, false);
  acc[1] = __builtin_amdgcn_wmma_f32_16x16x32_bf16(false, f.a, false, f.b1, (short)0, acc[1], false, false);
  acc[2] = __builtin_amdgcn_wmma_f32_16x16x32_bf16(false, f.a, false, f.b2, (short)0, acc[2], false, false);
  acc[3] = __builtin_amdgcn_wmma_f32_16x16x32_bf16(false, f.a, false, f.b3, (short)0, acc[3], false, false);
}

// ---------------------------------------------------------------------------
// Generic skinny GEMM: C[M,N] = epi(A1·B1ᵀ (+ A2·B2ᵀ) + bias1 + bias2)
// A: [M,K] bf16, B(=W): [N,K] bf16. Each wave computes a 16x64 tile via
// 4 accumulators of v_wmma_f32_16x16x32_bf16, with a depth-2 software
// pipeline (fragments for K-step i+1 in flight while WMMAs consume step i;
// partial s_wait_loadcnt instead of full drains) plus cacheline prefetch
// two K-steps ahead. 8 waves / block. Requires K % 64 == 0 (K = 1024/2048).
// ---------------------------------------------------------------------------
template<bool DUAL, bool TANH>
__global__ __launch_bounds__(256)
void gemm_bf16_wmma(const bf16s* __restrict__ A1, const bf16s* __restrict__ B1,
                    const bf16s* __restrict__ A2, const bf16s* __restrict__ B2,
                    const float* __restrict__ bias1, const float* __restrict__ bias2,
                    int M, int N, int K,
                    float* __restrict__ Cf, bf16s* __restrict__ Cbf, int ldc)
{
  const int tid  = threadIdx.x;
  const int lane = tid & 31;
  const int g    = lane >> 4;
  const int ln   = lane & 15;

  const int tilesN = (N + 63) >> 6;
  const int total  = (M >> 4) * tilesN;
  const int wid    = blockIdx.x * 8 + (tid >> 5);
  if (wid >= total) return;

  const int tm = wid / tilesN, tn = wid % tilesN;
  const int m0 = tm << 4, n0 = tn << 6;
  const int ma = m0 + ln;

  int nc[4];
#pragma unroll
  for (int i = 0; i < 4; ++i) {
    int n = n0 + 16 * i + ln;
    nc[i] = (n < N) ? n : (N - 1);                 // clamp OOB columns (guarded on store)
  }

  v8f acc[4] = {};

  // prologue: stage 0 in flight
  FragSet s0  = load_set(A1, B1, K, ma, nc, 0, g);
  FragSet s0d;
  if (DUAL) s0d = load_set(A2, B2, K, ma, nc, 0, g);

  for (int kk = 0; kk < K; kk += 64) {
    const int kb = kk + 32;                         // second half of this iter
    const int ka = (kk + 64 < K) ? kk + 64 : 0;     // next iter (wrap on last: dead loads)

    // prefetch two K-steps ahead (one cacheline covers 64 bf16 = 2 K-steps)
    if (kk + 128 < K) {
      prefetch_set(A1, B1, K, ma, nc, kk + 128);
      if (DUAL) prefetch_set(A2, B2, K, ma, nc, kk + 128);
    }

    // issue stage-1 loads, then consume stage 0
    FragSet s1  = load_set(A1, B1, K, ma, nc, kb, g);
    FragSet s1d;
    if (DUAL) s1d = load_set(A2, B2, K, ma, nc, kb, g);
    mma_set(acc, s0);
    if (DUAL) mma_set(acc, s0d);

    // issue next stage-0 loads, then consume stage 1
    s0 = load_set(A1, B1, K, ma, nc, ka, g);
    if (DUAL) s0d = load_set(A2, B2, K, ma, nc, ka, g);
    mma_set(acc, s1);
    if (DUAL) mma_set(acc, s1d);
  }

  // C layout: lane L element j -> m = m0 + j + 8*g, n = n0 + 16*i + (L&15)
#pragma unroll
  for (int i = 0; i < 4; ++i) {
    int n = n0 + 16 * i + ln;
    if (n < N) {
      float bb = 0.f;
      if (bias1) bb += bias1[n];
      if (bias2) bb += bias2[n];
#pragma unroll
      for (int j = 0; j < 8; ++j) {
        int m = m0 + j + 8 * g;
        float val = acc[i][j] + bb;
        if (TANH) val = tanhf(val);
        size_t o = (size_t)m * ldc + n;
        if (Cf)  Cf[o]  = val;
        if (Cbf) Cbf[o] = f2bf(val);
      }
    }
  }
}

// ---------------------------------------------------------------------------
// Elementwise / small kernels
// ---------------------------------------------------------------------------
__global__ void f2bf_kernel(const float* __restrict__ src, bf16s* __restrict__ dst, int n) {
  int i = blockIdx.x * blockDim.x + threadIdx.x;
  if (i < n) dst[i] = f2bf(src[i]);
}

__global__ void mask_kernel(const float* __restrict__ roi, float* __restrict__ mask) {
  int idx = blockIdx.x * blockDim.x + threadIdx.x;
  if (idx >= B_ * R_) return;
  const float* r = roi + (size_t)idx * H_;
  float s = 0.f;
  for (int i = 0; i < H_; ++i) s += r[i];
  mask[idx] = (s != 0.f) ? 0.f : -INFINITY;
}

// x = h1 = bf16(embed[SOS]); c0 = c1 = 0
__global__ void set_start_kernel(const float* __restrict__ embed,
                                 bf16s* __restrict__ xbf, bf16s* __restrict__ h1bf,
                                 float* __restrict__ c0, float* __restrict__ c1) {
  int idx = blockIdx.x * blockDim.x + threadIdx.x;   // B_*H_ threads
  if (idx >= B_ * H_) return;
  int h = idx & (H_ - 1);
  bf16s e = f2bf(embed[h]);                           // row SOS==0
  xbf[idx] = e; h1bf[idx] = e;
  c0[idx] = 0.f; c1[idx] = 0.f;
}

// h0 = h1 = attn_sos (taken from first half of xcat)
__global__ void set_h_kernel(const bf16s* __restrict__ xcat,
                             bf16s* __restrict__ h0bf, bf16s* __restrict__ h1bf) {
  int idx = blockIdx.x * blockDim.x + threadIdx.x;
  if (idx >= B_ * H_) return;
  int b = idx >> 10, h = idx & (H_ - 1);
  bf16s v = xcat[(size_t)b * H2_ + h];
  h0bf[idx] = v; h1bf[idx] = v;
}

// gates [B,4H] (i|f|g|o), c in/out f32, h out bf16
__global__ void lstm_pw_kernel(const float* __restrict__ gmat,
                               float* __restrict__ c, bf16s* __restrict__ hbf) {
  int idx = blockIdx.x * blockDim.x + threadIdx.x;   // B_*H_ threads
  if (idx >= B_ * H_) return;
  int b = idx >> 10, h = idx & (H_ - 1);
  const float* gr = gmat + (size_t)b * H4_;
  float gi = sigmoidf_(gr[h]);
  float gf = sigmoidf_(gr[H_ + h]);
  float gg = tanhf(gr[2 * H_ + h]);
  float go = sigmoidf_(gr[3 * H_ + h]);
  float c2 = gf * c[idx] + gi * gg;
  c[idx] = c2;
  hbf[idx] = f2bf(go * tanhf(c2));
}

// One block per batch row: scores -> softmax -> +mask -> ctx; writes xcat=[ctx|h1]
__global__ __launch_bounds__(256)
void attn_kernel(const bf16s* __restrict__ kbf, const float* __restrict__ vmat,
                 const bf16s* __restrict__ qbf, const float* __restrict__ mask,
                 const bf16s* __restrict__ h1bf, bf16s* __restrict__ xcat)
{
  __shared__ float s[64];
  const int b = blockIdx.x;
  const int tid = threadIdx.x, lane = tid & 31, wave = tid >> 5;

  const unsigned* qrow = (const unsigned*)(qbf + (size_t)b * H_);
  for (int r = wave; r < R_; r += 8) {
    const unsigned* krow = (const unsigned*)(kbf + ((size_t)b * R_ + r) * H_);
    float acc = 0.f;
    for (int i = lane; i < H_ / 2; i += 32) {      // 2 bf16 per dword, coalesced
      unsigned kp = krow[i], qp = qrow[i];
      acc += bf2f((bf16s)(kp & 0xFFFF)) * bf2f((bf16s)(qp & 0xFFFF));
      acc += bf2f((bf16s)(kp >> 16))    * bf2f((bf16s)(qp >> 16));
    }
    for (int off = 16; off; off >>= 1) acc += __shfl_xor(acc, off, 32);
    if (lane == 0) s[r] = acc;
  }
  __syncthreads();
  if (tid == 0) {
    float mx = -INFINITY;
    for (int r = 0; r < R_; ++r) mx = fmaxf(mx, s[r]);
    float sum = 0.f;
    for (int r = 0; r < R_; ++r) { float e = expf(s[r] - mx); s[r] = e; sum += e; }
    float inv = 1.f / sum;
    for (int r = 0; r < R_; ++r) s[r] = s[r] * inv + mask[b * R_ + r]; // mask AFTER softmax
  }
  __syncthreads();
  for (int h = tid; h < H_; h += 256) {
    const float* vb = vmat + (size_t)b * R_ * H_ + h;
    float acc = 0.f;
    for (int r = 0; r < R_; ++r) acc += s[r] * vb[(size_t)r * H_];
    xcat[(size_t)b * H2_ + h]      = f2bf(acc);
    xcat[(size_t)b * H2_ + H_ + h] = h1bf[(size_t)b * H_ + h];
  }
}

// Copy logits to d_out[B,V,T] slice t, argmax over V, gather embed[amax] -> x_bf
__global__ __launch_bounds__(256)
void argmax_store_kernel(const float* __restrict__ logits, const float* __restrict__ embed,
                         float* __restrict__ out, bf16s* __restrict__ xbf,
                         int t, int write_x)
{
  __shared__ float sv[256];
  __shared__ int   si[256];
  const int b = blockIdx.x, tid = threadIdx.x;
  const float* lr = logits + (size_t)b * V_;
  float best = -INFINITY; int bi = 0;
  for (int v = tid; v < V_; v += 256) {
    float x = lr[v];
    out[((size_t)b * V_ + v) * T_ + t] = x;
    if (x > best) { best = x; bi = v; }
  }
  sv[tid] = best; si[tid] = bi;
  __syncthreads();
  for (int stp = 128; stp; stp >>= 1) {
    if (tid < stp) {
      if (sv[tid + stp] > sv[tid] ||
          (sv[tid + stp] == sv[tid] && si[tid + stp] < si[tid])) {
        sv[tid] = sv[tid + stp]; si[tid] = si[tid + stp];
      }
    }
    __syncthreads();
  }
  if (write_x) {
    const float* er = embed + (size_t)si[0] * H_;
    for (int h = tid; h < H_; h += 256)
      xbf[(size_t)b * H_ + h] = f2bf(er[h]);
  }
}

// ---------------------------------------------------------------------------
// Host orchestration
// ---------------------------------------------------------------------------
static void launch_gemm(hipStream_t s,
                        const bf16s* A1, const bf16s* B1,
                        const bf16s* A2, const bf16s* B2,
                        const float* bias1, const float* bias2,
                        int M, int N, int K,
                        float* Cf, bf16s* Cbf, int ldc, int epi_tanh) {
  int tilesN = (N + 63) / 64;
  int total  = (M / 16) * tilesN;
  int blocks = (total + 7) / 8;
  if (A2) {
    gemm_bf16_wmma<true, false><<<blocks, 256, 0, s>>>(
        A1, B1, A2, B2, bias1, bias2, M, N, K, Cf, Cbf, ldc);
  } else if (epi_tanh) {
    gemm_bf16_wmma<false, true><<<blocks, 256, 0, s>>>(
        A1, B1, nullptr, nullptr, bias1, bias2, M, N, K, Cf, Cbf, ldc);
  } else {
    gemm_bf16_wmma<false, false><<<blocks, 256, 0, s>>>(
        A1, B1, nullptr, nullptr, bias1, bias2, M, N, K, Cf, Cbf, ldc);
  }
}

extern "C" void kernel_launch(void* const* d_in, const int* in_sizes, int n_in,
                              void* d_out, int out_size, void* d_ws, size_t ws_size,
                              hipStream_t stream) {
  (void)in_sizes; (void)n_in; (void)out_size; (void)ws_size;

  const float* roi   = (const float*)d_in[0];
  const float* embed = (const float*)d_in[1];
  const float* Wq    = (const float*)d_in[2];
  const float* bq    = (const float*)d_in[3];
  const float* Wk    = (const float*)d_in[4];
  const float* bk    = (const float*)d_in[5];
  const float* Wv    = (const float*)d_in[6];
  const float* bv    = (const float*)d_in[7];
  const float* Wih0  = (const float*)d_in[8];
  const float* Whh0  = (const float*)d_in[9];
  const float* bih0  = (const float*)d_in[10];
  const float* bhh0  = (const float*)d_in[11];
  const float* Wih1  = (const float*)d_in[12];
  const float* Whh1  = (const float*)d_in[13];
  const float* bih1  = (const float*)d_in[14];
  const float* bhh1  = (const float*)d_in[15];
  const float* projW = (const float*)d_in[16];
  const float* projb = (const float*)d_in[17];
  float* out = (float*)d_out;

  // workspace carve-up (256B aligned)
  char* base = (char*)d_ws;
  size_t off = 0;
  auto alloc = [&](size_t bytes) -> char* {
    char* p = base + off;
    off += (bytes + 255) & ~(size_t)255;
    return p;
  };
  bf16s* wq_bf   = (bf16s*)alloc((size_t)H_ * H_ * 2);
  bf16s* wk_bf   = (bf16s*)alloc((size_t)H_ * H_ * 2);
  bf16s* wv_bf   = (bf16s*)alloc((size_t)H_ * H_ * 2);
  bf16s* wih0_bf = (bf16s*)alloc((size_t)H4_ * H_ * 2);
  bf16s* whh0_bf = (bf16s*)alloc((size_t)H4_ * H_ * 2);
  bf16s* wih1_bf = (bf16s*)alloc((size_t)H4_ * H_ * 2);
  bf16s* whh1_bf = (bf16s*)alloc((size_t)H4_ * H_ * 2);
  bf16s* projW_bf= (bf16s*)alloc((size_t)V_ * H2_ * 2);
  bf16s* roi_bf  = (bf16s*)alloc((size_t)B_ * R_ * H_ * 2);
  bf16s* k_bf    = (bf16s*)alloc((size_t)B_ * R_ * H_ * 2);
  float* v_f     = (float*)alloc((size_t)B_ * R_ * H_ * 4);
  float* mask    = (float*)alloc((size_t)B_ * R_ * 4);
  bf16s* q_bf    = (bf16s*)alloc((size_t)B_ * H_ * 2);
  bf16s* x_bf    = (bf16s*)alloc((size_t)B_ * H_ * 2);
  bf16s* h0_bf   = (bf16s*)alloc((size_t)B_ * H_ * 2);
  bf16s* h1_bf   = (bf16s*)alloc((size_t)B_ * H_ * 2);
  float* c0_f    = (float*)alloc((size_t)B_ * H_ * 4);
  float* c1_f    = (float*)alloc((size_t)B_ * H_ * 4);
  float* g0_f    = (float*)alloc((size_t)B_ * H4_ * 4);
  float* g1_f    = (float*)alloc((size_t)B_ * H4_ * 4);
  bf16s* xcat_bf = (bf16s*)alloc((size_t)B_ * H2_ * 2);
  float* logits  = (float*)alloc((size_t)B_ * V_ * 4);

  auto conv = [&](const float* src, bf16s* dst, size_t n) {
    f2bf_kernel<<<(int)((n + 255) / 256), 256, 0, stream>>>(src, dst, (int)n);
  };

  // one-time: weight conversion
  conv(Wq,    wq_bf,   (size_t)H_ * H_);
  conv(Wk,    wk_bf,   (size_t)H_ * H_);
  conv(Wv,    wv_bf,   (size_t)H_ * H_);
  conv(Wih0,  wih0_bf, (size_t)H4_ * H_);
  conv(Whh0,  whh0_bf, (size_t)H4_ * H_);
  conv(Wih1,  wih1_bf, (size_t)H4_ * H_);
  conv(Whh1,  whh1_bf, (size_t)H4_ * H_);
  conv(projW, projW_bf,(size_t)V_ * H2_);
  conv(roi,   roi_bf,  (size_t)B_ * R_ * H_);

  mask_kernel<<<(B_ * R_ + 255) / 256, 256, 0, stream>>>(roi, mask);
  set_start_kernel<<<(B_ * H_) / 256, 256, 0, stream>>>(embed, x_bf, h1_bf, c0_f, c1_f);

  // k = tanh(roi·Wkᵀ + bk) (bf16), v = tanh(roi·Wvᵀ + bv) (f32)
  launch_gemm(stream, roi_bf, wk_bf, nullptr, nullptr, bk, nullptr,
              B_ * R_, H_, H_, nullptr, k_bf, H_, 1);
  launch_gemm(stream, roi_bf, wv_bf, nullptr, nullptr, bv, nullptr,
              B_ * R_, H_, H_, v_f, nullptr, H_, 1);

  // ---- SOS phase: attn(start_emb) and first logits (t = 0) ----
  launch_gemm(stream, h1_bf, wq_bf, nullptr, nullptr, bq, nullptr,
              B_, H_, H_, nullptr, q_bf, H_, 1);
  attn_kernel<<<B_, 256, 0, stream>>>(k_bf, v_f, q_bf, mask, h1_bf, xcat_bf);
  launch_gemm(stream, xcat_bf, projW_bf, nullptr, nullptr, projb, nullptr,
              B_, V_, H2_, logits, nullptr, V_, 0);
  argmax_store_kernel<<<B_, 256, 0, stream>>>(logits, embed, out, x_bf, 0, 0);
  // h0 = h1 = attn_sos ; x stays start_emb (per reference scan init)
  set_h_kernel<<<(B_ * H_) / 256, 256, 0, stream>>>(xcat_bf, h0_bf, h1_bf);

  // ---- recurrent loop, t = 1..31 ----
  for (int t = 1; t < T_; ++t) {
    // layer 0 gates: x·Wih0ᵀ + h0·Whh0ᵀ + biases
    launch_gemm(stream, x_bf, wih0_bf, h0_bf, whh0_bf, bih0, bhh0,
                B_, H4_, H_, g0_f, nullptr, H4_, 0);
    lstm_pw_kernel<<<(B_ * H_) / 256, 256, 0, stream>>>(g0_f, c0_f, h0_bf);
    // layer 1 gates: h0'·Wih1ᵀ + h1·Whh1ᵀ
    launch_gemm(stream, h0_bf, wih1_bf, h1_bf, whh1_bf, bih1, bhh1,
                B_, H4_, H_, g1_f, nullptr, H4_, 0);
    lstm_pw_kernel<<<(B_ * H_) / 256, 256, 0, stream>>>(g1_f, c1_f, h1_bf);
    // q = tanh(h1'·Wqᵀ + bq)
    launch_gemm(stream, h1_bf, wq_bf, nullptr, nullptr, bq, nullptr,
                B_, H_, H_, nullptr, q_bf, H_, 1);
    // attention -> xcat = [ctx | h1']
    attn_kernel<<<B_, 256, 0, stream>>>(k_bf, v_f, q_bf, mask, h1_bf, xcat_bf);
    // logits = xcat·projWᵀ + projb
    launch_gemm(stream, xcat_bf, projW_bf, nullptr, nullptr, projb, nullptr,
                B_, V_, H2_, logits, nullptr, V_, 0);
    // store slice t, argmax, x = embed[argmax]
    argmax_store_kernel<<<B_, 256, 0, stream>>>(logits, embed, out, x_bf, t, 1);
  }
}
